// StochasticRewiring_27049704030260
// MI455X (gfx1250) — matrix operand order repaired and checked by reading, not compiled, not run on hardware
//
#include <hip/hip_runtime.h>
#include <hip/hip_bf16.h>

typedef __attribute__((ext_vector_type(16))) _Float16       v16h;
typedef __attribute__((ext_vector_type(8)))  _Float16       v8h;
typedef __attribute__((ext_vector_type(8)))  unsigned short v8u;
typedef __attribute__((ext_vector_type(8)))  float          v8f;

#define DIM   128
#define HID   64
#define WAVES 4          // 4 waves / block = 128 threads, one 16-edge tile per wave
#define EPSV  1e-10f

// packed weight fragments in d_ws: frag f -> 32 lanes x 16 f16 (32B per lane)
//   f = kc*4 + nt            for W1 (kc 0..15, nt 0..3)   -> frags 0..63
//   f = 64 + kc*4 + nt       for W2 (kc 0..1,  nt 0..3)   -> frags 64..71
#define NFRAG_W1 64
#define NFRAG    72

__device__ __forceinline__ float sigmoidf_(float x) {
    return 1.0f / (1.0f + __expf(-x));
}

__device__ __forceinline__ v16h cat8(v8h a, v8h b) {
    return __builtin_shufflevector(a, b, 0, 1, 2, 3, 4, 5, 6, 7,
                                         8, 9, 10, 11, 12, 13, 14, 15);
}

__device__ __forceinline__ v8h habs8(v8h x) {           // clear sign bits == fabs
    const v8u mask = {0x7fff, 0x7fff, 0x7fff, 0x7fff, 0x7fff, 0x7fff, 0x7fff, 0x7fff};
    v8u t = __builtin_bit_cast(v8u, x);
    t = t & mask;
    return __builtin_bit_cast(v8h, t);
}

// one-time: convert W1/W2 (f32, row-major [K, 64]) into per-lane B-fragment
// layout, f16.  B frag (32x16): lane L -> col = L&15, K = 8*(L>>4)+t (t<8), 16+8*(L>>4)+(t-8) (t>=8).
__global__ void pack_weights_kernel(const float* __restrict__ W1,
                                    const float* __restrict__ W2,
                                    _Float16* __restrict__ ws) {
    const int f    = blockIdx.x;          // 0..71
    const int lane = threadIdx.x;         // 0..31
    const int half = lane >> 4;
    const int m    = lane & 15;

    const float* W;
    int kc, nt;
    if (f < NFRAG_W1) { W = W1; kc = f >> 2;              nt = f & 3; }
    else              { W = W2; kc = (f - NFRAG_W1) >> 2; nt = (f - NFRAG_W1) & 3; }

    const int n  = nt * 16 + m;
    const int kb = kc * 32 + half * 8;
    _Float16* dst = ws + (size_t)f * 512 + lane * 16;
#pragma unroll
    for (int t = 0; t < 16; ++t)
        dst[t] = (_Float16)W[(kb + (t < 8 ? t : t + 8)) * HID + n];
}

__global__ __launch_bounds__(32 * WAVES)
void edge_gate_mlp_kernel(const float* __restrict__ h,
                          const long long* __restrict__ ei,   // [2][E] int64
                          const float* __restrict__ u,
                          const float* __restrict__ b1,
                          const float* __restrict__ b2,
                          const float* __restrict__ W3, const float* __restrict__ b3,
                          const _Float16* __restrict__ ws,    // packed W1/W2 frags
                          float* __restrict__ out, int E, int tiles)
{
    __shared__ _Float16 sh_h[WAVES][2][16][DIM + 8];  // 272B rows: 16B aligned
    __shared__ _Float16 sh_x[WAVES][16][HID + 8];     // 144B rows: 16B aligned
    __shared__ float    sh_p[WAVES][16];

    const int lane = threadIdx.x & 31;
    const int wave = threadIdx.x >> 5;
    const int half = lane >> 4;
    const int m    = lane & 15;

    const int tile = blockIdx.x * WAVES + wave;
    if (tile >= tiles) return;
    const int ebase = tile * 16;

    const v16h* __restrict__ BW = (const v16h*)ws;    // frag f at BW[f*32 + lane]

    // ---------------- gather: hi/hj rows -> LDS (f32 -> f16) ----------------
#pragma unroll 1
    for (int r = 0; r < 16; ++r) {
        int e = ebase + r; if (e >= E) e = E - 1;
        const long long ri = ei[e];
        const long long ci = ei[(long long)E + e];
        const float4 hv = *(const float4*)(h + (size_t)ri * DIM + lane * 4);
        const float4 jv = *(const float4*)(h + (size_t)ci * DIM + lane * 4);
        _Float16* di = &sh_h[wave][0][r][lane * 4];
        _Float16* dj = &sh_h[wave][1][r][lane * 4];
        di[0] = (_Float16)hv.x; di[1] = (_Float16)hv.y;
        di[2] = (_Float16)hv.z; di[3] = (_Float16)hv.w;
        dj[0] = (_Float16)jv.x; dj[1] = (_Float16)jv.y;
        dj[2] = (_Float16)jv.z; dj[3] = (_Float16)jv.w;
    }
    asm volatile("s_wait_dscnt 0x0" ::: "memory");    // in-wave LDS visibility

    const _Float16 (*hi)[DIM + 8] = sh_h[wave][0];
    const _Float16 (*hj)[DIM + 8] = sh_h[wave][1];

    // ------- layer 1: feats[16,512] @ W1[512,64]  (64 WMMA per tile) --------
    // K chunk kc = seg*4 + cc; for fixed cc all 4 segments reuse the same
    // hi/hj LDS data -> 4 ds_load_b128 feed 16 WMMAs.
    v8f acc[4] = {};
#pragma unroll 1
    for (int cc = 0; cc < 4; ++cc) {
        const int b0 = cc * 32 + half * 8;            // 16B-aligned f16 offset
        const v8h ilo = *(const v8h*)&hi[m][b0];
        const v8h ihi = *(const v8h*)&hi[m][b0 + 16];
        const v8h jlo = *(const v8h*)&hj[m][b0];
        const v8h jhi = *(const v8h*)&hj[m][b0 + 16];

        v16h A[4];
        A[0] = cat8(ilo, ihi);                        // hi
        A[1] = cat8(jlo, jhi);                        // hj
        A[2] = cat8(habs8(ilo - jlo), habs8(ihi - jhi)); // |hi-hj|  (v_pk ops)
        A[3] = cat8(ilo * jlo, ihi * jhi);            // hi*hj

#pragma unroll
        for (int s = 0; s < 4; ++s) {
            const int kc = s * 4 + cc;
#pragma unroll
            for (int nt = 0; nt < 4; ++nt) {
                const v16h b = BW[(size_t)(kc * 4 + nt) * 32 + lane];
                acc[nt] = __builtin_amdgcn_wmma_f32_16x16x32_f16(
                    false, A[s], false, b, (short)0, acc[nt], false, false);
            }
        }
    }

    // bias + relu -> LDS (f16). C layout: VGPR r = row r+8*half, col = nt*16+m
#pragma unroll
    for (int nt = 0; nt < 4; ++nt) {
        const int n = nt * 16 + m;
        const float bb = b1[n];
#pragma unroll
        for (int r = 0; r < 8; ++r) {
            float v = acc[nt][r] + bb;
            v = v > 0.0f ? v : 0.0f;
            sh_x[wave][8 * half + r][n] = (_Float16)v;
        }
    }
    asm volatile("s_wait_dscnt 0x0" ::: "memory");

    // ------- layer 2: x[16,64] @ W2[64,64]  (8 WMMA per tile) ---------------
    v8f acc2[4] = {};
#pragma unroll
    for (int kc = 0; kc < 2; ++kc) {
        const int b0 = kc * 32 + half * 8;
        const v8h alo = *(const v8h*)&sh_x[wave][m][b0];
        const v8h ahi = *(const v8h*)&sh_x[wave][m][b0 + 16];
        const v16h a  = cat8(alo, ahi);
#pragma unroll
        for (int nt = 0; nt < 4; ++nt) {
            const v16h b = BW[(size_t)(NFRAG_W1 + kc * 4 + nt) * 32 + lane];
            acc2[nt] = __builtin_amdgcn_wmma_f32_16x16x32_f16(
                false, a, false, b, (short)0, acc2[nt], false, false);
        }
    }

    // bias + relu -> LDS (overwrite sh_x; prior reads already in registers)
#pragma unroll
    for (int nt = 0; nt < 4; ++nt) {
        const int n = nt * 16 + m;
        const float bb = b2[n];
#pragma unroll
        for (int r = 0; r < 8; ++r) {
            float v = acc2[nt][r] + bb;
            v = v > 0.0f ? v : 0.0f;
            sh_x[wave][8 * half + r][n] = (_Float16)v;
        }
    }
    asm volatile("s_wait_dscnt 0x0" ::: "memory");

    // ------- layer 3 + concrete-Bernoulli gating ----------------------------
    if (half == 0) {                      // lanes 0..15, one edge each
        float logit = b3[0];
#pragma unroll
        for (int n = 0; n < HID; ++n)
            logit += (float)sh_x[wave][m][n] * W3[n];

        const int e = ebase + m;
        float p = 0.0f;
        if (e < E) {
            p = sigmoidf_(logit);                           // TEMP = 1
            float uu = u[e];
            uu = fminf(fmaxf(uu, EPSV), 1.0f - EPSV);
            const float g    = logf(uu) - log1pf(-uu);
            const float gate = sigmoidf_(logit + g);
            out[e]             = gate;                      // new_edge_weight
            out[(size_t)E + e] = p;                         // probs
        }
        sh_p[wave][m] = p;
    }
    asm volatile("s_wait_dscnt 0x0" ::: "memory");

    if (lane == 0) {                      // per-wave partial -> global scalars
        float s = 0.0f;
#pragma unroll
        for (int r = 0; r < 16; ++r) s += sh_p[wave][r];
        atomicAdd(out + 2 * (size_t)E + 1, s);              // expected_num_edges
        atomicAdd(out + 2 * (size_t)E,     s / (float)E);   // expected_keep_rate
    }
}

__global__ void zero_scalars_kernel(float* __restrict__ out, size_t off) {
    out[off]     = 0.0f;
    out[off + 1] = 0.0f;
}

extern "C" void kernel_launch(void* const* d_in, const int* in_sizes, int n_in,
                              void* d_out, int out_size, void* d_ws, size_t ws_size,
                              hipStream_t stream) {
    const float*     h  = (const float*)d_in[0];
    const long long* ei = (const long long*)d_in[1];   // int64 edge_index [2][E]
    const float*     u  = (const float*)d_in[2];
    const float*     W1 = (const float*)d_in[3];
    const float*     b1 = (const float*)d_in[4];
    const float*     W2 = (const float*)d_in[5];
    const float*     b2 = (const float*)d_in[6];
    const float*     W3 = (const float*)d_in[7];
    const float*     b3 = (const float*)d_in[8];
    float*     out = (float*)d_out;
    _Float16*  ws  = (_Float16*)d_ws;                  // 72 frags * 1KB = 72KB

    const int E      = in_sizes[2];                    // u is [E]
    const int tiles  = (E + 15) / 16;
    const int blocks = (tiles + WAVES - 1) / WAVES;

    // one-time weight pre-pack (f32 -> f16, B-fragment layout) into scratch
    pack_weights_kernel<<<NFRAG, 32, 0, stream>>>(W1, W2, ws);
    // harness poisons d_out; scalar accumulators must start at zero
    zero_scalars_kernel<<<1, 1, 0, stream>>>(out, (size_t)2 * E);
    edge_gate_mlp_kernel<<<blocks, 32 * WAVES, 0, stream>>>(
        h, ei, u, b1, b2, W3, b3, ws, out, E, tiles);
}
